// TLSTM_19507741458805
// MI455X (gfx1250) — matrix operand order repaired and checked by reading, not compiled, not run on hardware
//
#include <hip/hip_runtime.h>
#include <hip/hip_bf16.h>

// ---------------------------------------------------------------------------
// Types for CDNA5 WMMA (wave32): v_wmma_f32_16x16x32_bf16
// ---------------------------------------------------------------------------
typedef __attribute__((ext_vector_type(16))) __bf16         v16bf;
typedef __attribute__((ext_vector_type(8)))  float          v8f;
typedef __attribute__((ext_vector_type(16))) unsigned short us16;
typedef __attribute__((ext_vector_type(8)))  unsigned short us8;

union UF16 { us16 v; unsigned short s[16]; v16bf b; };
union UF8  { us8  v; unsigned short s[8]; };

__device__ __forceinline__ unsigned short f2bf(float f) {
  unsigned u = __float_as_uint(f);
  u += 0x7FFFu + ((u >> 16) & 1u);          // round-to-nearest-even
  return (unsigned short)(u >> 16);
}
__device__ __forceinline__ float bf2f(unsigned short s) {
  return __uint_as_float(((unsigned)s) << 16);
}
__device__ __forceinline__ float fsig(float x) {
  return 1.0f / (1.0f + __expf(-x));
}
__device__ __forceinline__ float ftanh(float x) {
#if __has_builtin(__builtin_amdgcn_tanhf)
  return __builtin_amdgcn_tanhf(x);          // v_tanh_f32 (gfx1250)
#else
  float xc = fminf(fmaxf(x, -15.0f), 15.0f);
  float e  = __expf(2.0f * xc);
  return (e - 1.0f) / (e + 1.0f);
#endif
}
__device__ __forceinline__ v8f wmma_bf(v16bf a, v16bf b, v8f c) {
  return __builtin_amdgcn_wmma_f32_16x16x32_bf16(
      /*neg_a=*/false, a, /*neg_b=*/false, b,
      /*c_mod=*/(short)0, c, /*reuse_a=*/false, /*reuse_b=*/false);
}

// A-fragment (16x32 bf16, M rows in lanes) from row-major bf16, ld = 256.
// lane<16: K halves {0..7,16..23}; lane>=16: {8..15,24..31} (ISA A layout).
__device__ __forceinline__ v16bf load_afrag(const unsigned short* rowbase, int off) {
  us8 lo = *(const us8*)(rowbase + off);
  us8 hi = *(const us8*)(rowbase + off + 16);
  UF16 u;
  u.v = __builtin_shufflevector(lo, hi, 0,1,2,3,4,5,6,7,8,9,10,11,12,13,14,15);
  return u.b;
}
// B-fragment from pre-swizzled fragment-major weights: one aligned 32B load.
__device__ __forceinline__ v16bf load_bfrag(const unsigned short* frag, int nt, int ks, int lane) {
  UF16 u;
  u.v = *(const us16*)(frag + ((((size_t)nt * 8 + ks) * 32 + lane) << 4));
  return u.b;
}

// ---------------------------------------------------------------------------
// Workspace layout (bytes)
// Xg stored FRAGMENT-MAJOR: [t][m(8)][nt(64)][lane(32)][r(8)] bf16 so that the
// consumer's C-tile init is a single aligned b128 load per tile.
// ---------------------------------------------------------------------------
static constexpr size_t N_XG    = 1024ull * 8 * 64 * 32 * 8;       // = 128M bf16 elements
static constexpr size_t OFF_XG   = 0;
static constexpr size_t OFF_WCAT = OFF_XG   + N_XG * 2;            // 64 ntiles * 8 ks * 32 * 16 bf16
static constexpr size_t OFF_UR   = OFF_WCAT + 64ull  * 8 * 32 * 16 * 2;  // 80 ntiles (4 gates + decomp)
static constexpr size_t OFF_TD   = OFF_UR   + 80ull  * 8 * 32 * 16 * 2;  // T_decay fp32 [t][b]
static constexpr size_t OFF_HBF  = OFF_TD   + 1024ull * 128 * 4;   // h bf16, 2 banks of 128x256
static constexpr size_t OFF_CBF  = OFF_HBF  + 2ull * 32768 * 2;    // c bf16, 2 banks
static constexpr size_t OFF_CNT  = OFF_CBF  + 2ull * 32768 * 2;    // barrier counter

#define NWG 16

// ---------------------------------------------------------------------------
// Kernel 0: swizzle weights to B-fragment layout, zero state + barrier.
// B-frag element (lane L, half h) = W[ks*32 + 16*(h/8) + 8*(L/16) + h%8][nt*16 + L%16]
// ---------------------------------------------------------------------------
__global__ __launch_bounds__(256) void k_prep(
    const float* Wi, const float* Wf, const float* Wo, const float* Wc,
    const float* Ui, const float* Uf, const float* Uo, const float* Uc,
    const float* Wd,
    unsigned short* wcat, unsigned short* ur,
    unsigned short* hbf, unsigned short* cbf, unsigned* cnt)
{
  int idx = blockIdx.x * 256 + threadIdx.x;
  const int NW = 64 * 8 * 32 * 16;   // 262144
  const int NU = 80 * 8 * 32 * 16;   // 327680
  if (idx < NW) {
    int nt = idx >> 12, ks = (idx >> 9) & 7, L = (idx >> 4) & 31, hh = idx & 15;
    int k   = ks * 32 + ((hh >> 3) << 4) + ((L >> 4) << 3) + (hh & 7);
    int g   = nt >> 4;
    int col = ((nt & 15) << 4) + (L & 15);
    const float* W = (g == 0) ? Wi : (g == 1) ? Wf : (g == 2) ? Wo : Wc;
    wcat[idx] = f2bf(W[k * 256 + col]);
    return;
  }
  idx -= NW;
  if (idx < NU) {
    int nt = idx >> 12, ks = (idx >> 9) & 7, L = (idx >> 4) & 31, hh = idx & 15;
    int k = ks * 32 + ((hh >> 3) << 4) + ((L >> 4) << 3) + (hh & 7);
    float v;
    if (nt < 64) {
      int g   = nt >> 4;
      int col = ((nt & 15) << 4) + (L & 15);
      const float* U = (g == 0) ? Ui : (g == 1) ? Uf : (g == 2) ? Uo : Uc;
      v = U[k * 256 + col];
    } else {
      int col = ((nt - 64) << 4) + (L & 15);
      v = Wd[k * 256 + col];
    }
    ur[idx] = f2bf(v);
    return;
  }
  idx -= NU;
  if (idx < 65536) { hbf[idx] = 0; return; }
  idx -= 65536;
  if (idx < 65536) { cbf[idx] = 0; return; }
  idx -= 65536;
  if (idx == 0) *cnt = 0u;
}

// ---------------------------------------------------------------------------
// Kernel 1: Xg = x @ [Wi|Wf|Wo|Wc] + biases  (bf16 fragment-major, WMMA bf16)
// One workgroup (8 waves) per timestep t: wave m owns rows 16m..16m+15.
// Also computes T_decay[t][b] = 1/log(time/10 + e).
// ---------------------------------------------------------------------------
__global__ __launch_bounds__(256) void k_xgates(
    const float* __restrict__ inp,
    const float* __restrict__ bi, const float* __restrict__ bfv,
    const float* __restrict__ bo, const float* __restrict__ bc,
    const unsigned short* __restrict__ wcat,
    unsigned short* __restrict__ xg, float* __restrict__ td)
{
  int t    = blockIdx.x;
  int tid  = threadIdx.x;
  int lane = tid & 31, m = tid >> 5;
  int lrow = lane & 15, kofs = (lane >> 4) << 3;

  if (tid < 128) {
    float tm = inp[((size_t)tid * 1024 + t) * 257];
    td[t * 128 + tid] = 1.0f / logf(tm / 10.0f + 2.7183f);
  }

  int rowb = m * 16 + lrow;
  const float* xrow = inp + ((size_t)rowb * 1024 + t) * 257 + 1;

  v16bf a[8];
#pragma unroll
  for (int ks = 0; ks < 8; ++ks) {
    UF16 u;
    int base = ks * 32 + kofs;
#pragma unroll
    for (int h = 0; h < 8; ++h) {
      u.s[h]     = f2bf(xrow[base + h]);
      u.s[h + 8] = f2bf(xrow[base + 16 + h]);
    }
    a[ks] = u.b;
  }

  // fragment-major output base for this (t, m): [t][m][nt][lane][r]
  unsigned short* xgb = xg + ((((size_t)t * 8 + m) * 64) * 32 + lane) * 8;
  for (int nt = 0; nt < 64; ++nt) {
    int g = nt >> 4;
    const float* bias = (g == 0) ? bi : (g == 1) ? bfv : (g == 2) ? bo : bc;
    float bv = bias[((nt & 15) << 4) + lrow];
    v8f acc = {bv, bv, bv, bv, bv, bv, bv, bv};
#pragma unroll
    for (int ks = 0; ks < 8; ++ks)
      acc = wmma_bf(a[ks], load_bfrag(wcat, nt, ks, lane), acc);
    UF8 o;
#pragma unroll
    for (int r = 0; r < 8; ++r) o.s[r] = f2bf(acc[r]);
    *(us8*)(xgb + (size_t)nt * 32 * 8) = o.v;   // one coalesced b128 store
  }
}

// ---------------------------------------------------------------------------
// Kernel 2: persistent recurrence. NWG=16 workgroups (one per WGP); WG w owns
// u in [16w,16w+16) for all 4 gates + decomp -> elementwise is fully local.
// fp32 cell state c lives in REGISTERS for the whole scan (owner-private);
// bf16 h/c copies are double-buffered in global for the WMMA A-operands.
// One device-scope barrier per step (monotonic counter).
// ---------------------------------------------------------------------------
__global__ __launch_bounds__(256) void k_recur(
    const float* __restrict__ bdec,
    const unsigned short* __restrict__ xg,
    const unsigned short* __restrict__ ur,
    const float* __restrict__ td,
    unsigned short* hbf, unsigned short* cbf,
    unsigned* cnt, float* __restrict__ out)
{
  int w    = blockIdx.x;
  int tid  = threadIdx.x;
  int lane = tid & 31, m = tid >> 5;
  int lrow = lane & 15, kofs = (lane >> 4) << 3;
  int rowb = m * 16 + lrow;

  int u = (w << 4) + lrow;              // owned u column
  float bd = bdec[u];

  v8f cst;                               // fp32 cell state, owner-private
#pragma unroll
  for (int r = 0; r < 8; ++r) cst[r] = 0.0f;

  for (int t = 0; t < 1024; ++t) {
    const unsigned short* hr = hbf + ((t & 1) << 15);
    const unsigned short* cr = cbf + ((t & 1) << 15);
    unsigned short* hw = hbf + (((t + 1) & 1) << 15);
    unsigned short* cw = cbf + (((t + 1) & 1) << 15);

    // ---- A fragments from h_prev (bf16, row-major ld=256) ----
    v16bf a[8];
    const unsigned short* hrow = hr + rowb * 256;
#pragma unroll
    for (int ks = 0; ks < 8; ++ks) a[ks] = load_afrag(hrow, ks * 32 + kofs);

    // fragment-major Xg base for this (t, m): [t][m][nt][lane][r]
    const unsigned short* xgb = xg + ((((size_t)t * 8 + m) * 64) * 32 + lane) * 8;

    // ---- gate tiles: C-init = precomputed x@W+b tile (one b128), + h@U ----
    v8f acc[4];
#pragma unroll
    for (int g = 0; g < 4; ++g) {
      int nt = (g << 4) + w;
      UF8 xi;
      xi.v = *(const us8*)(xgb + (size_t)nt * 32 * 8);
      v8f c;
#pragma unroll
      for (int r = 0; r < 8; ++r) c[r] = bf2f(xi.s[r]);
#pragma unroll
      for (int ks = 0; ks < 8; ++ks)
        c = wmma_bf(a[ks], load_bfrag(ur, nt, ks, lane), c);
      acc[g] = c;
    }

    // ---- decomp tile: A = c_prev (bf16), C-init = b_decomp ----
    const unsigned short* crow = cr + rowb * 256;
#pragma unroll
    for (int ks = 0; ks < 8; ++ks) a[ks] = load_afrag(crow, ks * 32 + kofs);
    v8f accd = {bd, bd, bd, bd, bd, bd, bd, bd};
    {
      int nt = 64 + w;
#pragma unroll
      for (int ks = 0; ks < 8; ++ks)
        accd = wmma_bf(a[ks], load_bfrag(ur, nt, ks, lane), accd);
    }

    // ---- prefetch next step's Xg fragments (global_prefetch_b8) ----
    if (t + 1 < 1024) {
      const unsigned short* xgn = xg + ((((size_t)(t + 1) * 8 + m) * 64) * 32 + lane) * 8;
#pragma unroll
      for (int g = 0; g < 4; ++g)
        __builtin_prefetch(xgn + (size_t)(((g << 4) + w)) * 32 * 8, 0, 3);
    }

    // ---- elementwise TLSTM update (owner-private per (b,u)) ----
#pragma unroll
    for (int r = 0; r < 8; ++r) {
      int b       = m * 16 + r + kofs;
      float cprev = cst[r];
      float CST   = ftanh(accd[r]);
      float Td    = td[t * 128 + b];
      float cadj  = cprev - CST + Td * CST;
      float iv    = fsig(acc[0][r]);
      float fv    = fsig(acc[1][r]);
      float ov    = fsig(acc[2][r]);
      float Cv    = ftanh(acc[3][r]);
      float Ct    = fv * cadj + iv * Cv;
      cst[r] = Ct;
      cw[b * 256 + u] = f2bf(Ct);
      float h = ov * ftanh(Ct);
      out[((size_t)b * 1024 + t) * 256 + u] = h;
      hw[b * 256 + u] = f2bf(h);
    }

    // ---- device-scope barrier: monotonic counter, one signal per WG ----
    __threadfence();
    __syncthreads();
    if (tid == 0) {
      atomicAdd(cnt, 1u);
      unsigned target = (unsigned)NWG * (unsigned)(t + 1);
      while (__hip_atomic_load(cnt, __ATOMIC_RELAXED, __HIP_MEMORY_SCOPE_AGENT) < target)
        __builtin_amdgcn_s_sleep(2);
    }
    __syncthreads();
    __threadfence();
  }
}

// ---------------------------------------------------------------------------
extern "C" void kernel_launch(void* const* d_in, const int* in_sizes, int n_in,
                              void* d_out, int out_size, void* d_ws, size_t ws_size,
                              hipStream_t stream) {
  const float* inp = (const float*)d_in[0];
  const float* Wi  = (const float*)d_in[1];
  const float* Ui  = (const float*)d_in[2];
  const float* bi  = (const float*)d_in[3];
  const float* Wf  = (const float*)d_in[4];
  const float* Uf  = (const float*)d_in[5];
  const float* bfv = (const float*)d_in[6];
  const float* Wo  = (const float*)d_in[7];
  const float* Uo  = (const float*)d_in[8];
  const float* bo  = (const float*)d_in[9];
  const float* Wc  = (const float*)d_in[10];
  const float* Uc  = (const float*)d_in[11];
  const float* bc  = (const float*)d_in[12];
  const float* Wd  = (const float*)d_in[13];
  const float* bd  = (const float*)d_in[14];

  char* ws = (char*)d_ws;
  unsigned short* xg   = (unsigned short*)(ws + OFF_XG);
  unsigned short* wcat = (unsigned short*)(ws + OFF_WCAT);
  unsigned short* ur   = (unsigned short*)(ws + OFF_UR);
  float*          tdp  = (float*)(ws + OFF_TD);
  unsigned short* hbf  = (unsigned short*)(ws + OFF_HBF);
  unsigned short* cbf  = (unsigned short*)(ws + OFF_CBF);
  unsigned*       cnt  = (unsigned*)(ws + OFF_CNT);
  float* out = (float*)d_out;

  // total prep items: 262144 + 327680 + 65536 + 65536 + 1 = 720897
  k_prep<<<(720897 + 255) / 256, 256, 0, stream>>>(Wi, Wf, Wo, Wc, Ui, Uf, Uo, Uc, Wd,
                                                   wcat, ur, hbf, cbf, cnt);
  k_xgates<<<1024, 256, 0, stream>>>(inp, bi, bfv, bo, bc, wcat, xg, tdp);
  k_recur<<<NWG, 256, 0, stream>>>(bd, xg, ur, tdp, hbf, cbf, cnt, out);
}
